// TargetedComm_33483565040158
// MI455X (gfx1250) — matrix operand order repaired and checked by reading, not compiled, not run on hardware
//
#include <hip/hip_runtime.h>
#include <hip/hip_bf16.h>
#include <math.h>

// ---------------------------------------------------------------------------
// TargetedComm attention on MI455X (gfx1250, wave32, WMMA 16x16x32 bf16)
//   d_out = m_bar (8192x512 f32) ++ comm_attn (8192x8192 f32)
//   d_ws  = Q (bf16, n-major) | K (bf16, n-major) | Vt (bf16, c-major) : 24 MB
// HBM budget: attn write 268MB + softmax r/w 536MB + pv read 268MB ~= 1.1 GB
// ---------------------------------------------------------------------------

#define NN    8192
#define INDIM 1024
#define CC    512
#define BK    32
#define LSTR  20   // LDS row stride in u32 (= 40 bf16; 32 data + 8 pad)

typedef __attribute__((ext_vector_type(16))) __bf16 v16bf;
typedef __attribute__((ext_vector_type(8)))  float  v8f;

union Frag { v16bf v; unsigned int u[8]; uint4 q[2]; };

__device__ __forceinline__ unsigned short f2bf(float x) {
    unsigned int u = __float_as_uint(x);
    u += 0x7fffu + ((u >> 16) & 1u);          // round-to-nearest-even
    return (unsigned short)(u >> 16);
}
__device__ __forceinline__ unsigned int pack_bf2(float a, float b) {
    return (unsigned int)f2bf(a) | ((unsigned int)f2bf(b) << 16);
}

__device__ __forceinline__ v8f wmma_bf16(const Frag& a, const Frag& b, v8f c) {
    return __builtin_amdgcn_wmma_f32_16x16x32_bf16(false, a.v, false, b.v,
                                                   (short)0, c, false, false);
}

// --- CDNA5 async copy: 16B global -> LDS, tracked by ASYNCcnt ---------------
__device__ __forceinline__ void async_ld16(const void* lds_ptr, const void* g) {
    // low 32 bits of a flat LDS pointer are the wave-relative LDS offset
    unsigned lds_off = (unsigned)(unsigned long long)lds_ptr;
    asm volatile("global_load_async_to_lds_b128 %0, %1, off"
                 :: "v"(lds_off), "v"(g) : "memory");
}
template <int N>
__device__ __forceinline__ void wait_asynccnt() {
    asm volatile("s_wait_asynccnt %0" :: "i"(N) : "memory");
}

// A fragment (16x32 bf16) from LDS tile stored row-major, stride LSTR u32.
__device__ __forceinline__ void load_afrag(Frag& a, const unsigned int* lds,
                                           int rowBase, int lane) {
    const int m  = lane & 15;
    const int kb = (lane >> 4) ? 8 : 0;       // pair offset (16 bf16)
    const int base = (rowBase + m) * LSTR;
#pragma unroll
    for (int i = 0; i < 8; ++i) {
        a.u[i]     = lds[base + kb + i];      // K 0..15 / 16..31
    }
}
// B fragment (32x16 bf16): column n contiguous in LDS row n (stride LSTR u32).
__device__ __forceinline__ void load_bfrag(Frag& b, const unsigned int* lds,
                                           int colBase, int lane) {
    const int n  = lane & 15;
    const int kb = (lane >> 4) ? 8 : 0;       // pair offset (16 bf16)
    const int base = (colBase + n) * LSTR + kb;
#pragma unroll
    for (int i = 0; i < 8; ++i) b.u[i] = lds[base + i];
}

// ---------------------------------------------------------------------------
// Kernel 1: QKV projection. 64x64 tile/block, 8 waves of 16x32.
// blockIdx.z: 0=Q (1/sqrt(512) folded in), 1=K, 2=V (stored TRANSPOSED Vt[c][n])
// ---------------------------------------------------------------------------
__global__ __launch_bounds__(256) void qkv_kernel(
    const float* __restrict__ u, const float* __restrict__ z,
    const float* __restrict__ Wq, const float* __restrict__ bq,
    const float* __restrict__ Wk, const float* __restrict__ bk,
    const float* __restrict__ Wv, const float* __restrict__ bv,
    unsigned short* __restrict__ Qb, unsigned short* __restrict__ Kb,
    unsigned short* __restrict__ Vt)
{
    __shared__ unsigned int ldsA[64 * LSTR];
    __shared__ unsigned int ldsB[64 * LSTR];

    const int which = blockIdx.z;
    const float* W    = (which == 0) ? Wq : (which == 1) ? Wk : Wv;
    const float* bias = (which == 0) ? bq : (which == 1) ? bk : bv;
    const float scale = (which == 0) ? 0.044194173824159216f : 1.0f;

    const int r0 = blockIdx.x * 64;
    const int c0 = blockIdx.y * 64;
    const int tid  = threadIdx.x;
    const int lane = tid & 31;
    const int w    = tid >> 5;
    const int wm   = w & 3;
    const int wn   = w >> 2;

    v8f acc0 = {}; v8f acc1 = {};

    for (int k0 = 0; k0 < INDIM; k0 += BK) {
        const float* src = (k0 < 512) ? u : z;   // 32-wide tiles never straddle
        const int kk = k0 & 511;
#pragma unroll
        for (int j = 0; j < 4; ++j) {
            int p = tid + j * 256;               // 1024 bf16-pairs
            int row = p >> 4, kp = p & 15;
            const float* g = src + (size_t)(r0 + row) * 512 + kk + 2 * kp;
            ldsA[row * LSTR + kp] = pack_bf2(g[0], g[1]);
        }
#pragma unroll
        for (int j = 0; j < 4; ++j) {
            int p = tid + j * 256;
            int row = p >> 4, kp = p & 15;
            const float* g = W + (size_t)(c0 + row) * INDIM + k0 + 2 * kp;
            ldsB[row * LSTR + kp] = pack_bf2(g[0], g[1]);
        }
        __syncthreads();
        Frag a, b0, b1;
        load_afrag(a, ldsA, 16 * wm, lane);
        load_bfrag(b0, ldsB, 32 * wn, lane);
        load_bfrag(b1, ldsB, 32 * wn + 16, lane);
        acc0 = wmma_bf16(a, b0, acc0);
        acc1 = wmma_bf16(a, b1, acc1);
        __syncthreads();
    }

    const int col0 = c0 + 32 * wn + (lane & 15);
    const int hi8  = (lane >> 4) * 8;
    const float bia0 = bias[col0], bia1 = bias[col0 + 16];
#pragma unroll
    for (int i = 0; i < 8; ++i) {
        int row = r0 + 16 * wm + i + hi8;
        unsigned short v0 = f2bf((acc0[i] + bia0) * scale);
        unsigned short v1 = f2bf((acc1[i] + bia1) * scale);
        if (which == 2) {                         // V transposed: Vt[c][n]
            Vt[(size_t)col0        * NN + row] = v0;
            Vt[(size_t)(col0 + 16) * NN + row] = v1;
        } else {
            unsigned short* dst = (which == 0) ? Qb : Kb;
            dst[(size_t)row * CC + col0]      = v0;
            dst[(size_t)row * CC + col0 + 16] = v1;
        }
    }
}

// ---------------------------------------------------------------------------
// Kernel 2: scores = Q.K^T, diag=-inf.  128x128 tile/block, 8 waves of 32x64
// (8 WMMA/wave/step). Double-buffered async global->LDS staging (ASYNCcnt).
// ---------------------------------------------------------------------------
#define SBM 128
#define STILE (SBM * LSTR)              // u32 per tile
#define SBUF  (2 * STILE)               // A + B per buffer

__device__ __forceinline__ void scores_issue(const unsigned int* q32,
                                             const unsigned int* k32,
                                             unsigned int* lds, int p,
                                             int r0, int c0, int tid, int it)
{
    const int k0 = it * BK;
    unsigned int* bufA = lds + p * SBUF;
    unsigned int* bufB = bufA + STILE;
#pragma unroll
    for (int j = 0; j < 2; ++j) {
        int lin = tid + j * 256;                 // 512 16B-chunks per tile
        int r = lin >> 2, ch = lin & 3;
        const unsigned int* gA = q32 + (size_t)(r0 + r) * (CC / 2) + (k0 >> 1) + ch * 4;
        const unsigned int* gB = k32 + (size_t)(c0 + r) * (CC / 2) + (k0 >> 1) + ch * 4;
        async_ld16(&bufA[r * LSTR + ch * 4], gA);
        async_ld16(&bufB[r * LSTR + ch * 4], gB);
    }
}

__global__ __launch_bounds__(256) void scores_kernel(
    const unsigned short* __restrict__ Qb, const unsigned short* __restrict__ Kb,
    float* __restrict__ attn)
{
    __shared__ unsigned int lds[2 * SBUF];       // 40 KB, ping-pong

    const int r0 = blockIdx.x * SBM;
    const int c0 = blockIdx.y * SBM;
    const int tid  = threadIdx.x;
    const int lane = tid & 31;
    const int w    = tid >> 5;
    const int wm   = w & 3;                      // 4 row groups of 32
    const int wn   = w >> 2;                     // 2 col groups of 64

    const unsigned int* q32 = (const unsigned int*)Qb;
    const unsigned int* k32 = (const unsigned int*)Kb;

    v8f acc[2][4];
#pragma unroll
    for (int mi = 0; mi < 2; ++mi)
#pragma unroll
        for (int nj = 0; nj < 4; ++nj) { v8f zv = {}; acc[mi][nj] = zv; }

    constexpr int NIT = CC / BK;                 // 16
    scores_issue(q32, k32, lds, 0, r0, c0, tid, 0);
    scores_issue(q32, k32, lds, 1, r0, c0, tid, 1);

    for (int it = 0; it < NIT; ++it) {
        if (it == NIT - 1) wait_asynccnt<0>();   // everything in flight done
        else               wait_asynccnt<4>();   // allow newest batch in flight
        __syncthreads();

        const unsigned int* bufA = lds + (it & 1) * SBUF;
        const unsigned int* bufB = bufA + STILE;
        Frag a0, a1, b;
        load_afrag(a0, bufA, 32 * wm,      lane);
        load_afrag(a1, bufA, 32 * wm + 16, lane);
#pragma unroll
        for (int nj = 0; nj < 4; ++nj) {
            load_bfrag(b, bufB, 64 * wn + 16 * nj, lane);
            acc[0][nj] = wmma_bf16(a0, b, acc[0][nj]);
            acc[1][nj] = wmma_bf16(a1, b, acc[1][nj]);
        }
        __syncthreads();                         // done reading this buffer
        if (it + 2 < NIT)
            scores_issue(q32, k32, lds, it & 1, r0, c0, tid, it + 2);
    }

    const int nlo = lane & 15;
    const int hi8 = (lane >> 4) * 8;
#pragma unroll
    for (int mi = 0; mi < 2; ++mi)
#pragma unroll
        for (int nj = 0; nj < 4; ++nj)
#pragma unroll
            for (int i = 0; i < 8; ++i) {
                int gr = r0 + 32 * wm + 16 * mi + i + hi8;
                int gc = c0 + 64 * wn + 16 * nj + nlo;
                float v = acc[mi][nj][i];
                if (gr == gc) v = -INFINITY;
                attn[(size_t)gr * NN + gc] = v;
            }
}

// ---------------------------------------------------------------------------
// Kernel 3: in-place row softmax (row in registers, LDS tree reductions).
// ---------------------------------------------------------------------------
__global__ __launch_bounds__(256) void softmax_kernel(float* __restrict__ attn)
{
    __shared__ float red[256];
    const int row = blockIdx.x;
    const int tid = threadIdx.x;
    float* p = attn + (size_t)row * NN;

    float4 v[8];
#pragma unroll
    for (int j = 0; j < 8; ++j) v[j] = ((const float4*)p)[tid + j * 256];

    float m = -INFINITY;
#pragma unroll
    for (int j = 0; j < 8; ++j)
        m = fmaxf(m, fmaxf(fmaxf(v[j].x, v[j].y), fmaxf(v[j].z, v[j].w)));
    red[tid] = m; __syncthreads();
    for (int s = 128; s > 0; s >>= 1) {
        if (tid < s) red[tid] = fmaxf(red[tid], red[tid + s]);
        __syncthreads();
    }
    const float rowmax = red[0]; __syncthreads();

    float sum = 0.f;
#pragma unroll
    for (int j = 0; j < 8; ++j) {
        v[j].x = __expf(v[j].x - rowmax); v[j].y = __expf(v[j].y - rowmax);
        v[j].z = __expf(v[j].z - rowmax); v[j].w = __expf(v[j].w - rowmax);
        sum += v[j].x + v[j].y + v[j].z + v[j].w;
    }
    red[tid] = sum; __syncthreads();
    for (int s = 128; s > 0; s >>= 1) {
        if (tid < s) red[tid] += red[tid + s];
        __syncthreads();
    }
    const float tot = red[0];
    const float inv = (tot > 0.f) ? 1.f / tot : 0.f;   // nan_to_num(0)
#pragma unroll
    for (int j = 0; j < 8; ++j) {
        v[j].x *= inv; v[j].y *= inv; v[j].z *= inv; v[j].w *= inv;
        ((float4*)p)[tid + j * 256] = v[j];
    }
}

// ---------------------------------------------------------------------------
// Kernel 4: m_bar = P.V.  Each block: 32 rows x full 512 cols -> P is read
// from HBM exactly once. 8 waves of 16x128 (8 WMMA/wave/step). B fragments
// are contiguous uint4 loads straight from L2-resident Vt[c][n].
// ---------------------------------------------------------------------------
__global__ __launch_bounds__(256) void pv_kernel(
    const float* __restrict__ attn, const unsigned short* __restrict__ Vt,
    float* __restrict__ out)
{
    __shared__ unsigned int ldsA[32 * LSTR];

    const int r0 = blockIdx.x * 32;
    const int tid  = threadIdx.x;
    const int lane = tid & 31;
    const int w    = tid >> 5;
    const int wm   = w & 1;                      // 2 row groups of 16
    const int wn   = w >> 1;                     // 4 col groups of 128

    const unsigned int* vt32 = (const unsigned int*)Vt;

    v8f acc[8];
#pragma unroll
    for (int nj = 0; nj < 8; ++nj) { v8f zv = {}; acc[nj] = zv; }

    const int n   = lane & 15;
    const int hiK = (lane >> 4) ? 8 : 0;         // B frag K-pair offset

    for (int k0 = 0; k0 < NN; k0 += BK) {
        // stage A: P 32 rows x 32 k, fp32 -> bf16 (each P element read once)
#pragma unroll
        for (int j = 0; j < 2; ++j) {
            int p = tid + j * 256;               // 512 pairs
            int row = p >> 4, kp = p & 15;
            const float* g = attn + (size_t)(r0 + row) * NN + k0 + 2 * kp;
            ldsA[row * LSTR + kp] = pack_bf2(g[0], g[1]);
        }
        if (k0 + BK < NN)                        // L2 prefetch of next P tile
            __builtin_prefetch(attn + (size_t)(r0 + (tid >> 3)) * NN + (k0 + BK) + (tid & 7) * 4, 0, 1);
        __syncthreads();

        Frag a;
        load_afrag(a, ldsA, 16 * wm, lane);
#pragma unroll
        for (int nj = 0; nj < 8; ++nj) {
            const int col = 128 * wn + 16 * nj + n;          // Vt row
            const uint4* bp = (const uint4*)(vt32 + (size_t)col * (NN / 2)
                                             + (k0 >> 1) + hiK);
            Frag b; b.q[0] = bp[0]; b.q[1] = bp[1];
            acc[nj] = wmma_bf16(a, b, acc[nj]);
        }
        __syncthreads();
    }

    const int hi8 = (lane >> 4) * 8;
#pragma unroll
    for (int nj = 0; nj < 8; ++nj)
#pragma unroll
        for (int i = 0; i < 8; ++i) {
            int row = r0 + 16 * wm + i + hi8;
            int col = 128 * wn + 16 * nj + n;
            out[(size_t)row * CC + col] = acc[nj][i];
        }
}

// ---------------------------------------------------------------------------
extern "C" void kernel_launch(void* const* d_in, const int* in_sizes, int n_in,
                              void* d_out, int out_size, void* d_ws, size_t ws_size,
                              hipStream_t stream)
{
    const float* u  = (const float*)d_in[0];
    const float* z  = (const float*)d_in[1];
    const float* Wk = (const float*)d_in[2];
    const float* bk = (const float*)d_in[3];
    const float* Wv = (const float*)d_in[4];
    const float* bv = (const float*)d_in[5];
    const float* Wq = (const float*)d_in[6];
    const float* bq = (const float*)d_in[7];

    float* m_bar = (float*)d_out;
    float* attn  = (float*)d_out + (size_t)NN * CC;

    unsigned short* Qb = (unsigned short*)d_ws;
    unsigned short* Kb = Qb + (size_t)NN * CC;
    unsigned short* Vt = Kb + (size_t)NN * CC;   // transposed: Vt[c][n]

    qkv_kernel<<<dim3(NN / 64, CC / 64, 3), 256, 0, stream>>>(
        u, z, Wq, bq, Wk, bk, Wv, bv, Qb, Kb, Vt);

    scores_kernel<<<dim3(NN / SBM, NN / SBM), 256, 0, stream>>>(Qb, Kb, attn);

    softmax_kernel<<<NN, 256, 0, stream>>>(attn);

    pv_kernel<<<NN / 32, 256, 0, stream>>>(attn, Vt, m_bar);
}
